// SimpleINN2D_86354612453799
// MI455X (gfx1250) — matrix operand order repaired and checked by reading, not compiled, not run on hardware
//
#include <hip/hip_runtime.h>
#include <math.h>

// ---------------------------------------------------------------------------
// Problem constants (match reference)
// ---------------------------------------------------------------------------
#define B_ROWS   8192
#define D_DIM    512
#define H_DIM    512
#define L1_DIM   256
#define NB_BLK   8

typedef __attribute__((ext_vector_type(16))) __bf16 v16bf;
typedef __attribute__((ext_vector_type(8)))  float  v8f;

#if defined(__has_builtin)
#if __has_builtin(__builtin_amdgcn_global_load_async_to_lds_b128) && \
    __has_builtin(__builtin_amdgcn_s_wait_asynccnt)
#define USE_ASYNC_LDS 1
#endif
#endif

#ifdef USE_ASYNC_LDS
typedef int v4i_t __attribute__((vector_size(16)));
typedef __attribute__((address_space(1))) v4i_t gv4i_t;   // global (AS1)
typedef __attribute__((address_space(3))) v4i_t sv4i_t;   // LDS (AS3)
#endif

__device__ __forceinline__ unsigned short f32_to_bf16(float f) {
    union { float f; unsigned int u; } c; c.f = f;
    unsigned int u = c.u;
    u += 0x7fffu + ((u >> 16) & 1u);          // round-to-nearest-even
    return (unsigned short)(u >> 16);
}

// Stage 16 bytes global -> LDS (async DMA on CDNA5, sync copy fallback)
__device__ __forceinline__ void stage16(const unsigned short* g, unsigned short* l) {
#ifdef USE_ASYNC_LDS
    __builtin_amdgcn_global_load_async_to_lds_b128(
        (gv4i_t*)g, (sv4i_t*)l, 0, 0);
#else
    *(uint4*)l = *(const uint4*)g;
#endif
}

template <int N>
__device__ __forceinline__ void wait_async() {
#ifdef USE_ASYNC_LDS
    __builtin_amdgcn_s_wait_asynccnt(N);
#endif
}

// ---------------------------------------------------------------------------
// Weight transpose + f32->bf16 convert:  W[g][K][N] -> Wt[g][N][K]
// ---------------------------------------------------------------------------
__global__ void wt_conv_kernel(const float* __restrict__ W,
                               unsigned short* __restrict__ Wt,
                               int K, int N, int total) {
    int i = blockIdx.x * 256 + threadIdx.x;
    if (i >= total) return;
    int kn  = K * N;
    int g   = i / kn;
    int rem = i - g * kn;
    int k   = rem / N;
    int n   = rem - k * N;
    Wt[(size_t)g * kn + (size_t)n * K + k] = f32_to_bf16(W[i]);
}

// ---------------------------------------------------------------------------
// Column permutation gather: xp[m][j] = x[m][perm[j]]; bf16 copy of x2 half.
// ---------------------------------------------------------------------------
__global__ void permute_kernel(const float* __restrict__ x,
                               const int* __restrict__ perm,
                               float* __restrict__ xp,
                               unsigned short* __restrict__ x2b) {
    int i = blockIdx.x * 256 + threadIdx.x;       // over B*D
    int m = i >> 9;
    int j = i & 511;
    float v = x[(size_t)m * D_DIM + perm[j]];
    xp[i] = v;
    if (j >= L1_DIM)
        x2b[(size_t)m * L1_DIM + (j - L1_DIM)] = f32_to_bf16(v);
}

// ---------------------------------------------------------------------------
// Coupling epilogue: y = exp(0.954*atan(s/1.5))*xs + t
// ---------------------------------------------------------------------------
__global__ void couple_kernel(const float* __restrict__ r,
                              const float* __restrict__ xp, int xoff,
                              float* __restrict__ ynext, int yoff,
                              unsigned short* __restrict__ yb) {
    int i = blockIdx.x * 256 + threadIdx.x;       // over B*L1
    int m = i >> 8;
    int j = i & 255;
    size_t rb = (size_t)m * D_DIM;
    float s  = r[rb + j];
    float t  = r[rb + L1_DIM + j];
    float xs = xp[rb + xoff + j];
    float e  = expf(1.5f * 0.636f * atanf(s * (1.0f / 1.5f)));
    float y  = e * xs + t;
    ynext[rb + yoff + j] = y;
    if (yb) yb[(size_t)m * L1_DIM + j] = f32_to_bf16(y);
}

// ---------------------------------------------------------------------------
// bf16 WMMA GEMM:  out(M x 512) = A(M x K) @ Wt(512 x K)^T + bias, opt ReLU
//   Block: 256 thr = 8 waves, tile 128(M) x 128(N), K-step 32.
//   Waves: 4 along M (wm), 2 along N (wn). Wave tile 32(M) x 64(N)
//   -> 2 A-frags x 4 B-frags -> 8x v_wmma_f32_16x16x32_bf16 per K-step.
//   Staging: async global->LDS DMA (ASYNCcnt), double buffered.
// ---------------------------------------------------------------------------
template <int K, bool RELU, bool OUTBF16>
__global__ __launch_bounds__(256) void gemm_bf16_kernel(
        const unsigned short* __restrict__ A,
        const unsigned short* __restrict__ Wt,
        const float* __restrict__ bias,
        void* __restrict__ out) {
    constexpr int BM = 128, BN = 128, ST = 40;    // ST: padded K-stride (bf16)
    constexpr int NK = K >> 5;
    __shared__ __attribute__((aligned(16))) unsigned short As[2][BM][ST];
    __shared__ __attribute__((aligned(16))) unsigned short Bs[2][BN][ST];

    const int tid  = threadIdx.x;
    const int lane = tid & 31, wid = tid >> 5;
    const int wm = wid & 3, wn = wid >> 2;
    const int lr = lane & 15, lh = lane >> 4;
    const int m0 = blockIdx.y * BM;
    const int n0 = blockIdx.x * BN;

    // staging map: 1024 16B segments per tile-pair (512 A + 512 B), 4/thread
    const int srow = tid >> 2;                    // 0..63
    const int sseg = tid & 3;                     // 16B segment within 64B row
    const unsigned short* Ag0 = A  + (size_t)(m0 + srow) * K + sseg * 8;
    const unsigned short* Ag1 = A  + (size_t)(m0 + srow + 64) * K + sseg * 8;
    const unsigned short* Bg0 = Wt + (size_t)(n0 + srow) * K + sseg * 8;
    const unsigned short* Bg1 = Wt + (size_t)(n0 + srow + 64) * K + sseg * 8;

    v8f acc[2][4] = {};

    // preload K-tile 0 into buffer 0
    stage16(Ag0, &As[0][srow][sseg * 8]);
    stage16(Ag1, &As[0][srow + 64][sseg * 8]);
    stage16(Bg0, &Bs[0][srow][sseg * 8]);
    stage16(Bg1, &Bs[0][srow + 64][sseg * 8]);

#pragma unroll 2
    for (int kt = 0; kt < NK; ++kt) {
        const int buf = kt & 1;
        if (kt + 1 < NK) {                        // async-prefetch next tile
            const int ko = (kt + 1) << 5;
            stage16(Ag0 + ko, &As[buf ^ 1][srow][sseg * 8]);
            stage16(Ag1 + ko, &As[buf ^ 1][srow + 64][sseg * 8]);
            stage16(Bg0 + ko, &Bs[buf ^ 1][srow][sseg * 8]);
            stage16(Bg1 + ko, &Bs[buf ^ 1][srow + 64][sseg * 8]);
            wait_async<4>();                      // tile kt complete (in-order)
        } else {
            wait_async<0>();
        }
        __syncthreads();                          // tile kt visible to all waves

        // A fragments: lane lr = row, K = lh*16 + i (contiguous)
        union Frag { v16bf v; uint4 q[2]; };
        Frag af[2];
#pragma unroll
        for (int i = 0; i < 2; ++i) {
            const unsigned short* ap = &As[buf][wm * 32 + i * 16 + lr][0];
            af[i].q[0] = *(const uint4*)(ap + lh * 16);
            af[i].q[1] = *(const uint4*)(ap + lh * 16 + 8);
        }
        // B fragments: lane lr = col, K = lh*16 + i (contiguous)
        Frag bfr[4];
#pragma unroll
        for (int t = 0; t < 4; ++t) {
            const unsigned short* bp = &Bs[buf][wn * 64 + t * 16 + lr][lh * 16];
            bfr[t].q[0] = *(const uint4*)(bp);
            bfr[t].q[1] = *(const uint4*)(bp + 8);
        }
#pragma unroll
        for (int i = 0; i < 2; ++i)
#pragma unroll
            for (int t = 0; t < 4; ++t)
                acc[i][t] = __builtin_amdgcn_wmma_f32_16x16x32_bf16(
                                false, af[i].v, false, bfr[t].v, (short)0,
                                acc[i][t], false, false);

        __syncthreads();                          // reads done before overwrite
    }

    // epilogue: C layout -> lane lr = col, row = lh*8 + e
#pragma unroll
    for (int t = 0; t < 4; ++t) {
        const int col = n0 + wn * 64 + t * 16 + lr;
        const float bb = bias[col];
#pragma unroll
        for (int i = 0; i < 2; ++i) {
            const int rowb = m0 + wm * 32 + i * 16 + lh * 8;
#pragma unroll
            for (int e = 0; e < 8; ++e) {
                float v = acc[i][t][e] + bb;
                if (RELU) v = v > 0.0f ? v : 0.0f;
                size_t idx = (size_t)(rowb + e) * 512 + col;
                if (OUTBF16) ((unsigned short*)out)[idx] = f32_to_bf16(v);
                else         ((float*)out)[idx] = v;
            }
        }
    }
}

// ---------------------------------------------------------------------------
// Host orchestration
// ---------------------------------------------------------------------------
extern "C" void kernel_launch(void* const* d_in, const int* in_sizes, int n_in,
                              void* d_out, int out_size, void* d_ws, size_t ws_size,
                              hipStream_t stream) {
    const float* x     = (const float*)d_in[0];
    const int*   perms = (const int*)d_in[1];
    const float* W1    = (const float*)d_in[2];
    const float* b1    = (const float*)d_in[3];
    const float* W2    = (const float*)d_in[4];
    const float* b2    = (const float*)d_in[5];
    float*       out   = (float*)d_out;

    char* ws = (char*)d_ws;
    size_t off = 0;
    auto alloc = [&](size_t bytes) -> void* {
        void* p = ws + off;
        off += (bytes + 255) & ~(size_t)255;
        return p;
    };
    const size_t w1e = (size_t)NB_BLK * 2 * L1_DIM * H_DIM;   // 2,097,152
    const size_t w2e = (size_t)NB_BLK * 2 * H_DIM * D_DIM;    // 4,194,304
    unsigned short* W1t  = (unsigned short*)alloc(w1e * 2);
    unsigned short* W2t  = (unsigned short*)alloc(w2e * 2);
    float*          xp   = (float*)alloc((size_t)B_ROWS * D_DIM * 4);
    float*          rbuf = (float*)alloc((size_t)B_ROWS * D_DIM * 4);
    unsigned short* hbuf = (unsigned short*)alloc((size_t)B_ROWS * D_DIM * 2);
    unsigned short* ab   = (unsigned short*)alloc((size_t)B_ROWS * L1_DIM * 2);
    float*          P0   = (float*)alloc((size_t)B_ROWS * D_DIM * 4);
    float*          P1   = (float*)alloc((size_t)B_ROWS * D_DIM * 4);

    // one-time (per call) weight transpose+convert
    wt_conv_kernel<<<(int)(w1e / 256), 256, 0, stream>>>(W1, W1t, L1_DIM, H_DIM, (int)w1e);
    wt_conv_kernel<<<(int)(w2e / 256), 256, 0, stream>>>(W2, W2t, H_DIM, D_DIM, (int)w2e);

    const dim3 ggrid(D_DIM / 128, B_ROWS / 128); // (4, 64)
    const int  pgrid = B_ROWS * D_DIM / 256;     // 16384
    const int  cgrid = B_ROWS * L1_DIM / 256;    // 8192

    const float* xcur = x;
    for (int k = 0; k < NB_BLK; ++k) {
        permute_kernel<<<pgrid, 256, 0, stream>>>(xcur, perms + k * D_DIM, xp, ab);

        float* xnext = (k == NB_BLK - 1) ? out : ((k & 1) ? P1 : P0);

        // half-step 2: r2 = MLP(x2; W[k,1]) ; y1 = e(s2)*x1 + t2
        int g = k * 2 + 1;
        gemm_bf16_kernel<L1_DIM, true,  true ><<<ggrid, 256, 0, stream>>>(
            ab,   W1t + (size_t)g * H_DIM * L1_DIM, b1 + (size_t)g * H_DIM, hbuf);
        gemm_bf16_kernel<H_DIM,  false, false><<<ggrid, 256, 0, stream>>>(
            hbuf, W2t + (size_t)g * D_DIM * H_DIM,  b2 + (size_t)g * D_DIM, rbuf);
        couple_kernel<<<cgrid, 256, 0, stream>>>(rbuf, xp, 0, xnext, 0, ab);

        // half-step 1: r1 = MLP(y1; W[k,0]) ; y2 = e(s1)*x2 + t1
        g = k * 2 + 0;
        gemm_bf16_kernel<L1_DIM, true,  true ><<<ggrid, 256, 0, stream>>>(
            ab,   W1t + (size_t)g * H_DIM * L1_DIM, b1 + (size_t)g * H_DIM, hbuf);
        gemm_bf16_kernel<H_DIM,  false, false><<<ggrid, 256, 0, stream>>>(
            hbuf, W2t + (size_t)g * D_DIM * H_DIM,  b2 + (size_t)g * D_DIM, rbuf);
        couple_kernel<<<cgrid, 256, 0, stream>>>(rbuf, xp, L1_DIM, xnext, L1_DIM, nullptr);

        xcur = xnext;
    }
}
